// SMAQQuantizer_17360257810704
// MI455X (gfx1250) — matrix operand order repaired and checked by reading, not compiled, not run on hardware
//
#include <hip/hip_runtime.h>

typedef __attribute__((ext_vector_type(16))) _Float16 v16h;
typedef __attribute__((ext_vector_type(8)))  _Float16 v8h;
typedef __attribute__((ext_vector_type(8)))  float    v8f;
typedef __attribute__((ext_vector_type(4)))  float    v4f;

#define DIM        128
#define LDS_STRIDE 136                    // halves per row: 128 + 8 pad (conflict-free b128 reads)
#define MAT_HALFS  (DIM * LDS_STRIDE)     // 17408 halves per 128x128 f16 matrix

__device__ __forceinline__ v16h cat16(v8h lo, v8h hi) {
  return __builtin_shufflevector(lo, hi, 0,1,2,3,4,5,6,7,8,9,10,11,12,13,14,15);
}

// split-f16 conversion of 8 f32 values: hi = rne(f16), lo = f16(v - (float)hi)
__device__ __forceinline__ void cvt8_split(v4f a, v4f b, v8h* hi, v8h* lo) {
  v8h h, l;
  #pragma unroll
  for (int i = 0; i < 4; ++i) {
    _Float16 x = (_Float16)a[i]; h[i]     = x; l[i]     = (_Float16)(a[i] - (float)x);
    _Float16 y = (_Float16)b[i]; h[i + 4] = y; l[i + 4] = (_Float16)(b[i] - (float)y);
  }
  *hi = h; *lo = l;
}

__device__ __forceinline__ float sq4(v4f a) {
  return a[0]*a[0] + a[1]*a[1] + a[2]*a[2] + a[3]*a[3];
}

// A-operand fragment (16x32 f16): lo-lanes hold K {0..7,16..23}, hi-lanes K {8..15,24..31}
__device__ __forceinline__ v16h loadFragA(const _Float16* base, int row, int c, int hi) {
  const _Float16* p = base + row * LDS_STRIDE + 32 * c + 8 * hi;
  return cat16(*(const v8h*)p, *(const v8h*)(p + 16));
}

// B-operand fragment (32x16 f16): lane = column; lo-lanes K {0..7,16..23}, hi-lanes K {8..15,24..31} of chunk
__device__ __forceinline__ v16h loadFragB(const _Float16* base, int row, int c, int hi) {
  const _Float16* p = base + row * LDS_STRIDE + 32 * c + 8 * hi;
  return cat16(*(const v8h*)p, *(const v8h*)(p + 16));
}

__device__ __forceinline__ v8f wmma_f16(v16h a, v16h b, v8f c) {
  return __builtin_amdgcn_wmma_f32_16x16x32_f16(false, a, false, b, (short)0, c, false, false);
}

__global__ __launch_bounds__(256)
void smaq_fused_kernel(const float* __restrict__ k,
                       const float* __restrict__ E,
                       const float* __restrict__ Einv,
                       const float* __restrict__ centroids,
                       const float* __restrict__ boundaries,
                       float* __restrict__ out,
                       int rows)
{
  extern __shared__ _Float16 smem[];
  _Float16* ldsEh = smem;                   // E high halves
  _Float16* ldsEl = smem + MAT_HALFS;       // E residual halves
  _Float16* ldsEi = smem + 2 * MAT_HALFS;   // E_inv (plain f16)

  const int tid    = threadIdx.x;
  const int w      = tid >> 5;
  const int lane   = tid & 31;
  const int laneLo = lane & 15;
  const int hi     = lane >> 4;

  _Float16* ldsY   = smem + 3 * MAT_HALFS + w * (16 * LDS_STRIDE);
  float*    ldsNrm = (float*)(smem + 4 * MAT_HALFS);   // 8 waves x 16 norms

  // ---- cooperative fill: E -> (hi,lo) split f16, E_inv -> f16 ----
  const float2* E2  = (const float2*)E;
  const float2* Ei2 = (const float2*)Einv;
  for (int t = tid; t < DIM * 64; t += 256) {
    int r  = t >> 6;
    int c2 = t & 63;
    float2 a = E2[t];
    float2 b = Ei2[t];
    int o = r * LDS_STRIDE + 2 * c2;
    _Float16 hx = (_Float16)a.x, hy = (_Float16)a.y;
    ldsEh[o] = hx;                          ldsEh[o + 1] = hy;
    ldsEl[o] = (_Float16)(a.x - (float)hx); ldsEl[o + 1] = (_Float16)(a.y - (float)hy);
    ldsEi[o] = (_Float16)b.x;               ldsEi[o + 1] = (_Float16)b.y;
  }

  float cent[8], bnd[7];
  #pragma unroll
  for (int i = 0; i < 8; ++i) cent[i] = centroids[i];
  #pragma unroll
  for (int i = 0; i < 7; ++i) bnd[i]  = boundaries[i];

  __syncthreads();

  const int nBlk = rows >> 7;
  for (int blk = blockIdx.x; blk < nBlk; blk += gridDim.x) {
    const int rowBase = (blk << 7) + (w << 4);
    if (rowBase >= rows) continue;          // wave-uniform: EXEC stays all-ones

    // ---- stream 16x128 k tile (non-temporal) into split A fragments + sum-of-squares ----
    const v4f* krow = (const v4f*)(k + (size_t)(rowBase + laneLo) * DIM);
    const int sOff = hi ? 2 : 0;
    float ss = 0.0f;
    v16h Ah[4], Al[4];
    #pragma unroll
    for (int c = 0; c < 4; ++c) {
      v4f f0 = __builtin_nontemporal_load(&krow[8 * c + sOff + 0]);
      v4f f1 = __builtin_nontemporal_load(&krow[8 * c + sOff + 1]);
      v4f f2 = __builtin_nontemporal_load(&krow[8 * c + sOff + 4]);
      v4f f3 = __builtin_nontemporal_load(&krow[8 * c + sOff + 5]);
      ss += sq4(f0) + sq4(f1) + sq4(f2) + sq4(f3);
      v8h h0, l0, h1, l1;
      cvt8_split(f0, f1, &h0, &l0);
      cvt8_split(f2, f3, &h1, &l1);
      Ah[c] = cat16(h0, h1);
      Al[c] = cat16(l0, l1);
    }
    ss += __shfl_xor(ss, 16, 32);
    float nrm = sqrtf(ss);
    if (!hi) ldsNrm[w * 16 + laneLo] = nrm;

    float nrmv[8], rnv[8];
    #pragma unroll
    for (int g = 0; g < 8; ++g) {
      nrmv[g] = ldsNrm[w * 16 + g + 8 * hi];
      rnv[g]  = __builtin_amdgcn_rcpf(nrmv[g] + 1e-10f);   // v_rcp_f32, no IEEE div chain
    }

    // ---- GEMM1 (compensated f16): y = (Ah+Al)@(Bh+Bl)^T * rn; quantize -> ldsY ----
    v16h Bh = loadFragB(ldsEh, laneLo, 0, hi);
    v16h Bl = loadFragB(ldsEl, laneLo, 0, hi);
    v8f acc = {};
    #pragma unroll
    for (int i = 0; i < 32; ++i) {
      const int nb = i >> 2, c = i & 3;
      v16h BhC = Bh, BlC = Bl;
      if (i < 31) {                          // prefetch next fragment before the WMMAs
        const int i2 = i + 1;
        Bh = loadFragB(ldsEh, (i2 >> 2) * 16 + laneLo, i2 & 3, hi);
        Bl = loadFragB(ldsEl, (i2 >> 2) * 16 + laneLo, i2 & 3, hi);
      }
      if (c == 0) acc = (v8f){};
      acc = wmma_f16(Al[c], BhC, acc);       // lo x hi
      acc = wmma_f16(Ah[c], BlC, acc);       // hi x lo
      acc = wmma_f16(Ah[c], BhC, acc);       // hi x hi (exact products, f32 accumulate)
      if (c == 3) {
        #pragma unroll
        for (int g = 0; g < 8; ++g) {
          float yv = acc[g] * rnv[g];
          float yh = cent[0];                // searchsorted(left): idx = #{b_j < y}
          #pragma unroll
          for (int j = 0; j < 7; ++j) yh = (yv > bnd[j]) ? cent[j + 1] : yh;
          ldsY[(g + 8 * hi) * LDS_STRIDE + nb * 16 + laneLo] = (_Float16)yh;
        }
      }
    }

    // ---- GEMM2: k_hat = y_hat @ Einv^T (plain f16); scale by norm, NT store ----
    v16h A2[4];
    #pragma unroll
    for (int c = 0; c < 4; ++c) A2[c] = loadFragA(ldsY, laneLo, c, hi);

    v16h B2 = loadFragB(ldsEi, laneLo, 0, hi);
    #pragma unroll
    for (int i = 0; i < 32; ++i) {
      const int jb = i >> 2, c = i & 3;
      v16h B2C = B2;
      if (i < 31) {
        const int i2 = i + 1;
        B2 = loadFragB(ldsEi, (i2 >> 2) * 16 + laneLo, i2 & 3, hi);
      }
      if (c == 0) acc = (v8f){};
      acc = wmma_f16(A2[c], B2C, acc);
      if (c == 3) {
        #pragma unroll
        for (int g = 0; g < 8; ++g) {
          float* dst = out + (size_t)(rowBase + g + 8 * hi) * DIM + jb * 16 + laneLo;
          __builtin_nontemporal_store(acc[g] * nrmv[g], dst);
        }
      }
    }
  }
}

extern "C" void kernel_launch(void* const* d_in, const int* in_sizes, int n_in,
                              void* d_out, int out_size, void* d_ws, size_t ws_size,
                              hipStream_t stream) {
  (void)n_in; (void)out_size; (void)d_ws; (void)ws_size;
  const float* k    = (const float*)d_in[0];
  const float* E    = (const float*)d_in[1];
  const float* Einv = (const float*)d_in[2];
  const float* cent = (const float*)d_in[3];
  const float* bnd  = (const float*)d_in[4];
  float* out = (float*)d_out;

  const long rows = (long)in_sizes[0] / DIM;    // 524288 for reference shapes
  int nBlk = (int)(rows >> 7);
  if (nBlk < 1) nBlk = 1;
  int grid = nBlk < 8192 ? nBlk : 8192;

  // LDS: Eh + El + Einv + 8 per-wave y_hat tiles (all f16, padded rows) + norms
  size_t shmem = (size_t)(4 * MAT_HALFS) * 2 + 8 * 16 * sizeof(float);

  smaq_fused_kernel<<<grid, 256, shmem, stream>>>(k, E, Einv, cent, bnd, out, (int)rows);
}
